// SparseMoEBlock_51616916963672
// MI455X (gfx1250) — matrix operand order repaired and checked by reading, not compiled, not run on hardware
//
#include <hip/hip_runtime.h>

#define EMB   1024
#define NEXP  8
#define NTOK  4096      // 2 * 2048 tokens
#define HID   1024      // MOE_HIDDEN
#define MAXTOK 4096     // per-expert capacity (worst case)
#define NSLOT (NTOK*2)  // total expert-token slots (top-2)

typedef __bf16 bf16;
typedef bf16  v16bf __attribute__((ext_vector_type(16)));
typedef bf16  v8bf  __attribute__((ext_vector_type(8)));
typedef float v8f   __attribute__((ext_vector_type(8)));

// ---------- helpers ----------
__device__ __forceinline__ unsigned short f32_to_bf16_rne(float f) {
  unsigned u = __float_as_uint(f);
  unsigned r = u + 0x7FFFu + ((u >> 16) & 1u);
  return (unsigned short)(r >> 16);
}
__device__ __forceinline__ float bf16bits_to_f32(unsigned short h) {
  return __uint_as_float(((unsigned)h) << 16);
}

// A fragment: 16x32 bf16 tile, lane holds 16 contiguous K (k = 16*laneHi + s), matching B.
// rowk = pointer to (row of A) + k0.  laneHi = lane>=16.
__device__ __forceinline__ v16bf load_a_frag(const unsigned short* rowk, int laneHi) {
  return *reinterpret_cast<const v16bf*>(rowk + (laneHi ? 16 : 0));
}
// B fragment: 32x16, B[k][n] = W[n][k]; lane n holds 16 contiguous K-halves.
__device__ __forceinline__ v16bf load_b_frag(const unsigned short* wrow_k0, int laneHi) {
  return *reinterpret_cast<const v16bf*>(wrow_k0 + (laneHi ? 16 : 0));
}
__device__ __forceinline__ v8f wmma_bf16(v16bf a, v16bf b, v8f c) {
  return __builtin_amdgcn_wmma_f32_16x16x32_bf16(false, a, false, b, (short)0, c, false, false);
}

// ---------- kernels ----------
__global__ void init_kernel(float* __restrict__ out, int n, int* __restrict__ counts) {
  int i = blockIdx.x * blockDim.x + threadIdx.x;
  if (i < n) out[i] = 0.0f;
  if (i < NEXP) counts[i] = 0;
}

__global__ void split_kernel(const float* __restrict__ src,
                             unsigned short* __restrict__ hi,
                             unsigned short* __restrict__ lo, int n) {
  int i = blockIdx.x * blockDim.x + threadIdx.x;
  if (i >= n) return;
  float v = src[i];
  unsigned short h = f32_to_bf16_rne(v);
  float fh = bf16bits_to_f32(h);
  unsigned short l = f32_to_bf16_rne(v - fh);
  hi[i] = h;
  lo[i] = l;
}

__global__ __launch_bounds__(256)
void router_kernel(const float* __restrict__ x, const float* __restrict__ rw,
                   const float* __restrict__ rb, int* __restrict__ counts,
                   int* __restrict__ tok_list, float* __restrict__ wt_list) {
  int wave = threadIdx.x >> 5;
  int lane = threadIdx.x & 31;
  int t = blockIdx.x * (blockDim.x >> 5) + wave;
  if (t >= NTOK) return;
  const float* h = x + (size_t)t * EMB;
  float acc[NEXP];
#pragma unroll
  for (int e = 0; e < NEXP; ++e) acc[e] = 0.0f;
  for (int k = lane; k < EMB; k += 32) {
    float hv = h[k];
#pragma unroll
    for (int e = 0; e < NEXP; ++e) acc[e] += hv * rw[e * EMB + k];
  }
#pragma unroll
  for (int e = 0; e < NEXP; ++e) {
#pragma unroll
    for (int off = 16; off > 0; off >>= 1)
      acc[e] += __shfl_xor(acc[e], off, 32);
  }
  if (lane == 0) {
    float s[NEXP];
#pragma unroll
    for (int e = 0; e < NEXP; ++e) s[e] = acc[e] + rb[e];
    int i0 = 0;
#pragma unroll
    for (int e = 1; e < NEXP; ++e) if (s[e] > s[i0]) i0 = e;
    int i1 = (i0 == 0) ? 1 : 0;
#pragma unroll
    for (int e = 0; e < NEXP; ++e) if (e != i0 && s[e] > s[i1]) i1 = e;
    float e1 = __expf(s[i1] - s[i0]);   // <= 1
    float inv = 1.0f / (1.0f + e1);
    float w0 = inv, w1 = e1 * inv;      // softmax over top-2
    int p0 = atomicAdd(&counts[i0], 1);
    tok_list[i0 * MAXTOK + p0] = t;  wt_list[i0 * MAXTOK + p0] = w0;
    int p1 = atomicAdd(&counts[i1], 1);
    tok_list[i1 * MAXTOK + p1] = t;  wt_list[i1 * MAXTOK + p1] = w1;
  }
}

__global__ void offsets_kernel(const int* __restrict__ counts, int* __restrict__ offs) {
  if (threadIdx.x == 0 && blockIdx.x == 0) {
    int a = 0;
    for (int e = 0; e < NEXP; ++e) { offs[e] = a; a += counts[e]; }
    offs[NEXP] = a;
  }
}

// GEMM1: for expert e, G = H_gathered @ gate_up_w[e]^T; swiglu(G) -> a (bf16 hi/lo)
// Each wave computes TWO 16(token) x 16(a-col) tiles (32 tokens) sharing B fragments,
// with separate gate/up accumulators per tile (3xBF16 split precision).
__global__ __launch_bounds__(256)
void gemm1_kernel(const unsigned short* __restrict__ h_hi,
                  const unsigned short* __restrict__ h_lo,
                  const unsigned short* __restrict__ w_hi,
                  const unsigned short* __restrict__ w_lo,
                  const float* __restrict__ gub,
                  const int* __restrict__ counts, const int* __restrict__ offs,
                  const int* __restrict__ tok_list,
                  unsigned short* __restrict__ a_hi,
                  unsigned short* __restrict__ a_lo) {
  int e = blockIdx.z;
  int cnt = counts[e];
  int m0base = blockIdx.x * 32;          // 32 rows per block tile
  if (m0base >= cnt) return;
  int wave = threadIdx.x >> 5;
  int lane = threadIdx.x & 31;
  int laneHi = lane >> 4;
  int n = lane & 15;

  int acol = (blockIdx.y * 8 + wave) * 16 + n;   // a-column for this lane (0..1023)
  int s0 = min(m0base + n, cnt - 1);             // A rows (clamped)
  int s1 = min(m0base + 16 + n, cnt - 1);
  int tok0 = tok_list[e * MAXTOK + s0];
  int tok1 = tok_list[e * MAXTOK + s1];

  const unsigned short* A0h = h_hi + (size_t)tok0 * EMB;
  const unsigned short* A0l = h_lo + (size_t)tok0 * EMB;
  const unsigned short* A1h = h_hi + (size_t)tok1 * EMB;
  const unsigned short* A1l = h_lo + (size_t)tok1 * EMB;
  size_t gr = ((size_t)e * 2 * HID + (size_t)2 * acol) * EMB;      // gate row
  size_t ur = gr + EMB;                                           // up row (2*acol+1)
  const unsigned short* Bg_hi = w_hi + gr;
  const unsigned short* Bg_lo = w_lo + gr;
  const unsigned short* Bu_hi = w_hi + ur;
  const unsigned short* Bu_lo = w_lo + ur;

  v8f accg[2] = {{}, {}};
  v8f accu[2] = {{}, {}};
  for (int k0 = 0; k0 < EMB; k0 += 32) {
    v16bf a0h = load_a_frag(A0h + k0, laneHi);
    v16bf a0l = load_a_frag(A0l + k0, laneHi);
    v16bf a1h = load_a_frag(A1h + k0, laneHi);
    v16bf a1l = load_a_frag(A1l + k0, laneHi);
    v16bf bgh = load_b_frag(Bg_hi + k0, laneHi);
    v16bf bgl = load_b_frag(Bg_lo + k0, laneHi);
    v16bf buh = load_b_frag(Bu_hi + k0, laneHi);
    v16bf bul = load_b_frag(Bu_lo + k0, laneHi);
    accg[0] = wmma_bf16(a0l, bgh, accg[0]);
    accg[0] = wmma_bf16(a0h, bgl, accg[0]);
    accg[0] = wmma_bf16(a0h, bgh, accg[0]);
    accu[0] = wmma_bf16(a0l, buh, accu[0]);
    accu[0] = wmma_bf16(a0h, bul, accu[0]);
    accu[0] = wmma_bf16(a0h, buh, accu[0]);
    accg[1] = wmma_bf16(a1l, bgh, accg[1]);
    accg[1] = wmma_bf16(a1h, bgl, accg[1]);
    accg[1] = wmma_bf16(a1h, bgh, accg[1]);
    accu[1] = wmma_bf16(a1l, buh, accu[1]);
    accu[1] = wmma_bf16(a1h, bul, accu[1]);
    accu[1] = wmma_bf16(a1h, buh, accu[1]);
  }

  float bg = gub[(size_t)e * 2 * HID + 2 * acol];
  float bu = gub[(size_t)e * 2 * HID + 2 * acol + 1];
  int base = offs[e];
#pragma unroll
  for (int t = 0; t < 2; ++t) {
#pragma unroll
    for (int r = 0; r < 8; ++r) {
      int m = t * 16 + laneHi * 8 + r;
      int slot = m0base + m;
      if (slot < cnt) {
        float g = accg[t][r] + bg;
        float u = accu[t][r] + bu;
        float gc = fminf(fmaxf(g, -7.0f), 7.0f);
        float uc = fminf(fmaxf(u, -7.0f), 7.0f);
        float sig = 1.0f / (1.0f + __expf(-1.702f * g));
        float a = gc * sig * (uc + 1.0f);
        unsigned short h16 = f32_to_bf16_rne(a);
        unsigned short l16 = f32_to_bf16_rne(a - bf16bits_to_f32(h16));
        size_t gi = (size_t)(base + slot) * HID + acol;
        a_hi[gi] = h16;
        a_lo[gi] = l16;
      }
    }
  }
}

// GEMM2: oe = a @ down_w[e]^T + down_b; out[token] += weight * oe.
// Two M-tiles per wave sharing B fragments; exactly 2 contributions per output
// element onto exact zero -> bitwise deterministic.
__global__ __launch_bounds__(256)
void gemm2_kernel(const unsigned short* __restrict__ a_hi,
                  const unsigned short* __restrict__ a_lo,
                  const unsigned short* __restrict__ dw_hi,
                  const unsigned short* __restrict__ dw_lo,
                  const float* __restrict__ db,
                  const int* __restrict__ counts, const int* __restrict__ offs,
                  const int* __restrict__ tok_list, const float* __restrict__ wt_list,
                  float* __restrict__ out) {
  int e = blockIdx.z;
  int cnt = counts[e];
  int m0base = blockIdx.x * 32;
  if (m0base >= cnt) return;
  int wave = threadIdx.x >> 5;
  int lane = threadIdx.x & 31;
  int laneHi = lane >> 4;
  int n = lane & 15;

  int dcol = (blockIdx.y * 8 + wave) * 16 + n;   // output emb column
  int base = offs[e];
  int s0 = min(m0base + n, cnt - 1);
  int s1 = min(m0base + 16 + n, cnt - 1);
  const unsigned short* A0h = a_hi + (size_t)(base + s0) * HID;
  const unsigned short* A0l = a_lo + (size_t)(base + s0) * HID;
  const unsigned short* A1h = a_hi + (size_t)(base + s1) * HID;
  const unsigned short* A1l = a_lo + (size_t)(base + s1) * HID;
  size_t wr = ((size_t)e * EMB + dcol) * HID;
  const unsigned short* Bhi = dw_hi + wr;
  const unsigned short* Blo = dw_lo + wr;

  v8f acc[2] = {{}, {}};
  for (int k0 = 0; k0 < HID; k0 += 32) {
    v16bf a0h = load_a_frag(A0h + k0, laneHi);
    v16bf a0l = load_a_frag(A0l + k0, laneHi);
    v16bf a1h = load_a_frag(A1h + k0, laneHi);
    v16bf a1l = load_a_frag(A1l + k0, laneHi);
    v16bf bh = load_b_frag(Bhi + k0, laneHi);
    v16bf bl = load_b_frag(Blo + k0, laneHi);
    acc[0] = wmma_bf16(a0l, bh, acc[0]);
    acc[0] = wmma_bf16(a0h, bl, acc[0]);
    acc[0] = wmma_bf16(a0h, bh, acc[0]);
    acc[1] = wmma_bf16(a1l, bh, acc[1]);
    acc[1] = wmma_bf16(a1h, bl, acc[1]);
    acc[1] = wmma_bf16(a1h, bh, acc[1]);
  }

  float bias = db[(size_t)e * EMB + dcol];
#pragma unroll
  for (int t = 0; t < 2; ++t) {
#pragma unroll
    for (int r = 0; r < 8; ++r) {
      int m = t * 16 + laneHi * 8 + r;
      int slot = m0base + m;
      if (slot < cnt) {
        int tok = tok_list[e * MAXTOK + slot];
        float w = wt_list[e * MAXTOK + slot];
        float val = (acc[t][r] + bias) * w;
        __hip_atomic_fetch_add(&out[(size_t)tok * EMB + dcol], val,
                               __ATOMIC_RELAXED, __HIP_MEMORY_SCOPE_AGENT);
      }
    }
  }
}

// ---------- launch ----------
extern "C" void kernel_launch(void* const* d_in, const int* in_sizes, int n_in,
                              void* d_out, int out_size, void* d_ws, size_t ws_size,
                              hipStream_t stream) {
  const float* x   = (const float*)d_in[0];
  const float* rw  = (const float*)d_in[1];
  const float* rb  = (const float*)d_in[2];
  const float* guw = (const float*)d_in[3];
  const float* gub = (const float*)d_in[4];
  const float* dw  = (const float*)d_in[5];
  const float* db  = (const float*)d_in[6];
  float* out = (float*)d_out;

  char* ws = (char*)d_ws;
  size_t o = 0;
  auto take = [&](size_t bytes) -> void* {
    void* p = (void*)(ws + o);
    o += (bytes + 255) & ~(size_t)255;
    return p;
  };
  int*   counts = (int*)take(NEXP * sizeof(int));
  int*   offs   = (int*)take((NEXP + 1) * sizeof(int));
  int*   tok_list = (int*)take((size_t)NEXP * MAXTOK * sizeof(int));
  float* wt_list  = (float*)take((size_t)NEXP * MAXTOK * sizeof(float));
  unsigned short* h_hi   = (unsigned short*)take((size_t)NTOK * EMB * 2);
  unsigned short* h_lo   = (unsigned short*)take((size_t)NTOK * EMB * 2);
  unsigned short* guw_hi = (unsigned short*)take((size_t)NEXP * 2 * HID * EMB * 2);
  unsigned short* guw_lo = (unsigned short*)take((size_t)NEXP * 2 * HID * EMB * 2);
  unsigned short* dw_hi  = (unsigned short*)take((size_t)NEXP * EMB * HID * 2);
  unsigned short* dw_lo  = (unsigned short*)take((size_t)NEXP * EMB * HID * 2);
  unsigned short* a_hi   = (unsigned short*)take((size_t)NSLOT * HID * 2);
  unsigned short* a_lo   = (unsigned short*)take((size_t)NSLOT * HID * 2);

  // 1) zero out + counters
  {
    int n = NTOK * EMB;
    init_kernel<<<(n + 255) / 256, 256, 0, stream>>>(out, n, counts);
  }
  // 2) fp32 -> bf16 hi/lo splits
  {
    int n = NTOK * EMB;
    split_kernel<<<(n + 255) / 256, 256, 0, stream>>>(x, h_hi, h_lo, n);
    n = NEXP * 2 * HID * EMB;
    split_kernel<<<(n + 255) / 256, 256, 0, stream>>>(guw, guw_hi, guw_lo, n);
    n = NEXP * EMB * HID;
    split_kernel<<<(n + 255) / 256, 256, 0, stream>>>(dw, dw_hi, dw_lo, n);
  }
  // 3) router (one wave32 per token)
  router_kernel<<<NTOK / 8, 256, 0, stream>>>(x, rw, rb, counts, tok_list, wt_list);
  // 4) prefix sum
  offsets_kernel<<<1, 32, 0, stream>>>(counts, offs);
  // 5) gate/up GEMM + swiglu  (32 tokens x 128 a-cols per block)
  {
    dim3 grid(MAXTOK / 32, HID / (16 * 8), NEXP);
    gemm1_kernel<<<grid, 256, 0, stream>>>(h_hi, h_lo, guw_hi, guw_lo, gub,
                                           counts, offs, tok_list, a_hi, a_lo);
  }
  // 6) down GEMM + weighted scatter
  {
    dim3 grid(MAXTOK / 32, EMB / (16 * 8), NEXP);
    gemm2_kernel<<<grid, 256, 0, stream>>>(a_hi, a_lo, dw_hi, dw_lo, db,
                                           counts, offs, tok_list, wt_list, out);
  }
}
